// Mesh3DActorCritic_55405078118895
// MI455X (gfx1250) — compile-verified, matrix operand order checked by reading, not run on hardware
//
#include <hip/hip_runtime.h>
#include <stdint.h>

// ---------------------------------------------------------------------------
// Problem constants (match reference)
// ---------------------------------------------------------------------------
static constexpr int Bc = 8;
static constexpr int Ec = 3000;
static constexpr int Fc = 2000;
static constexpr int Nc = Ec + Fc;   // 5000
static constexpr int Mc = 24000;
static constexpr int DH = 128;
static constexpr int Lc = 4;
#define NEG_INF (-1.0e9f)

typedef __attribute__((ext_vector_type(16))) _Float16 v16h;
typedef __attribute__((ext_vector_type(8)))  float    v8f;

// Fast SiLU: x * sigmoid(x) using the hardware reciprocal (v_rcp_f32, TRANS
// pipe, co-executes with WMMA) instead of the IEEE divide expansion.
__device__ __forceinline__ float siluf(float v) {
    return v * __builtin_amdgcn_rcpf(1.0f + __expf(-v));
}

// ---------------------------------------------------------------------------
// CDNA5 async copy: global f16 -> LDS, 16 bytes per lane, tracked by ASYNCcnt.
// Low 32 bits of a generic pointer to __shared__ are the LDS byte offset.
// ---------------------------------------------------------------------------
__device__ __forceinline__ void async_copy_b128(const _Float16* gptr, _Float16* lptr) {
    unsigned lds = (unsigned)(uintptr_t)(void*)lptr;
    unsigned long long ga = (unsigned long long)(uintptr_t)gptr;
    asm volatile("global_load_async_to_lds_b128 %0, %1, off"
                 :: "v"(lds), "v"(ga) : "memory");
}
__device__ __forceinline__ void wait_async() {
    asm volatile("s_wait_asynccnt 0x0" ::: "memory");
}

// ---------------------------------------------------------------------------
// WMMA fragment load: 16-bit A/B fragment per ISA layout.
// lane r (0-15): VGPR v<4 -> K = 2v, 2v+1 ; v>=4 -> K = 16+2(v-4), +1
// lane r+16    : same rows, K offset +8 (i.e. K in [8..15] and [24..31])
// p points at (row base + ktile*32) in a row-major f16 buffer.
// ---------------------------------------------------------------------------
__device__ __forceinline__ v16h load_frag(const _Float16* p, int half) {
    v16h f;
    const int base = half * 8;
#pragma unroll
    for (int v = 0; v < 8; ++v) {
        int kb = (v < 4 ? 2 * v : 16 + 2 * (v - 4)) + base;
        f[2 * v]     = p[kb];
        f[2 * v + 1] = p[kb + 1];
    }
    return f;
}

#define WMMA_F16(a, b, c) \
    __builtin_amdgcn_wmma_f32_16x16x32_f16(false, (a), false, (b), (short)0, (c), false, false)

// ---------------------------------------------------------------------------
// GEMM stages: 16 rows x 128 cols, blockDim=128 (4 waves, wave w -> cols [32w,32w+32))
// A: LDS f16, row stride lda. Wt: global f16 [128][ldw] (transposed, K padded).
// ---------------------------------------------------------------------------
__device__ __forceinline__ void gemm_f16(const _Float16* As, int lda, int kTiles,
                                         const _Float16* Wt, int ldw,
                                         const float* bias, bool act,
                                         _Float16* Out, int ldo) {
    const int tid = threadIdx.x;
    const int lane = tid & 31, wave = tid >> 5;
    const int n0 = wave * 32;
    const int r = lane & 15, half = lane >> 4;
    v8f d0 = {}, d1 = {};
    for (int kt = 0; kt < kTiles; ++kt) {
        v16h a  = load_frag(As + (size_t)r * lda + kt * 32, half);
        v16h b0 = load_frag(Wt + (size_t)(n0 + r) * ldw + kt * 32, half);
        v16h b1 = load_frag(Wt + (size_t)(n0 + 16 + r) * ldw + kt * 32, half);
        d0 = WMMA_F16(a, b0, d0);
        d1 = WMMA_F16(a, b1, d1);
    }
#pragma unroll
    for (int v = 0; v < 8; ++v) {
        int m = v + 8 * half;
        float e0 = d0[v] + bias[n0 + r];
        float e1 = d1[v] + bias[n0 + 16 + r];
        if (act) { e0 = siluf(e0); e1 = siluf(e1); }
        Out[(size_t)m * ldo + n0 + r]      = (_Float16)e0;
        Out[(size_t)m * ldo + n0 + 16 + r] = (_Float16)e1;
    }
}

__device__ __forceinline__ void gemm_f32(const _Float16* As, int lda, int kTiles,
                                         const _Float16* Wt, int ldw,
                                         const float* bias, bool act,
                                         const float* rowscale,
                                         float* OutF, int ldof,
                                         _Float16* OutH, int ldoh) {
    const int tid = threadIdx.x;
    const int lane = tid & 31, wave = tid >> 5;
    const int n0 = wave * 32;
    const int r = lane & 15, half = lane >> 4;
    v8f d0 = {}, d1 = {};
    for (int kt = 0; kt < kTiles; ++kt) {
        v16h a  = load_frag(As + (size_t)r * lda + kt * 32, half);
        v16h b0 = load_frag(Wt + (size_t)(n0 + r) * ldw + kt * 32, half);
        v16h b1 = load_frag(Wt + (size_t)(n0 + 16 + r) * ldw + kt * 32, half);
        d0 = WMMA_F16(a, b0, d0);
        d1 = WMMA_F16(a, b1, d1);
    }
#pragma unroll
    for (int v = 0; v < 8; ++v) {
        int m = v + 8 * half;
        float e0 = d0[v] + bias[n0 + r];
        float e1 = d1[v] + bias[n0 + 16 + r];
        if (act) { e0 = siluf(e0); e1 = siluf(e1); }
        if (rowscale) { float s = rowscale[m]; e0 *= s; e1 *= s; }
        OutF[(size_t)m * ldof + n0 + r]      = e0;
        OutF[(size_t)m * ldof + n0 + 16 + r] = e1;
        if (OutH) {
            OutH[(size_t)m * ldoh + n0 + r]      = (_Float16)e0;
            OutH[(size_t)m * ldoh + n0 + 16 + r] = (_Float16)e1;
        }
    }
}

// Global f32 output (stride 128) + optional global f16 mirror (stride 128).
__device__ __forceinline__ void gemm_global(const _Float16* As, int lda, int kTiles,
                                            const _Float16* Wt, int ldw,
                                            const float* bias, bool act,
                                            float* OutG, _Float16* OutH16, int validRows) {
    const int tid = threadIdx.x;
    const int lane = tid & 31, wave = tid >> 5;
    const int n0 = wave * 32;
    const int r = lane & 15, half = lane >> 4;
    v8f d0 = {}, d1 = {};
    for (int kt = 0; kt < kTiles; ++kt) {
        v16h a  = load_frag(As + (size_t)r * lda + kt * 32, half);
        v16h b0 = load_frag(Wt + (size_t)(n0 + r) * ldw + kt * 32, half);
        v16h b1 = load_frag(Wt + (size_t)(n0 + 16 + r) * ldw + kt * 32, half);
        d0 = WMMA_F16(a, b0, d0);
        d1 = WMMA_F16(a, b1, d1);
    }
#pragma unroll
    for (int v = 0; v < 8; ++v) {
        int m = v + 8 * half;
        if (m < validRows) {
            float e0 = d0[v] + bias[n0 + r];
            float e1 = d1[v] + bias[n0 + 16 + r];
            if (act) { e0 = siluf(e0); e1 = siluf(e1); }
            OutG[(size_t)m * 128 + n0 + r]      = e0;
            OutG[(size_t)m * 128 + n0 + 16 + r] = e1;
            if (OutH16) {
                OutH16[(size_t)m * 128 + n0 + r]      = (_Float16)e0;
                OutH16[(size_t)m * 128 + n0 + 16 + r] = (_Float16)e1;
            }
        }
    }
}

// ---------------------------------------------------------------------------
// Weight conversion: f32 W[K][N] -> f16 Wt[N][Kpad] (zero-padded K)
// ---------------------------------------------------------------------------
__global__ void convert_w(const float* __restrict__ W, _Float16* __restrict__ Wt,
                          int K, int Kpad, int Nout) {
    int i = blockIdx.x * blockDim.x + threadIdx.x;
    if (i >= Kpad * Nout) return;
    int n = i / Kpad, k = i % Kpad;
    Wt[i] = (k < K) ? (_Float16)W[(size_t)k * Nout + n] : (_Float16)0.0f;
}

// ---------------------------------------------------------------------------
// Embedding: h[b, off+r, :] = W1 @ silu(W0 @ feats + b0) + b1 (f32 + f16 mirror)
// ---------------------------------------------------------------------------
__global__ void embed_kernel(const float* __restrict__ feats, int rows, int tiles,
                             int nodeOffset,
                             const _Float16* W0, const float* b0,
                             const _Float16* W1, const float* b1,
                             float* __restrict__ h, _Float16* __restrict__ h16) {
    __shared__ _Float16 sA[16 * 48];
    __shared__ _Float16 sB[16 * 144];
    int b  = blockIdx.x / tiles;
    int r0 = (blockIdx.x % tiles) * 16;
    int valid = rows - r0; if (valid > 16) valid = 16;
    for (int idx = threadIdx.x; idx < 16 * 32; idx += 128) {
        int r = idx >> 5, c = idx & 31;
        float v = 0.0f;
        if (r < valid && c < 16) v = feats[((size_t)b * rows + r0 + r) * 16 + c];
        sA[r * 48 + c] = (_Float16)v;
    }
    __syncthreads();
    gemm_f16(sA, 48, 1, W0, 32, b0, true, sB, 144);
    __syncthreads();
    size_t rowBase = ((size_t)b * Nc + nodeOffset + r0) * 128;
    gemm_global(sB, 144, 4, W1, 128, b1, false, h + rowBase, h16 + rowBase, valid);
}

// ---------------------------------------------------------------------------
// Message kernel: one block = 16 edges of one batch.
// h_i/h_j gathered from the f16 mirror straight into LDS via async-to-LDS.
// phi_m (3 stages) -> m_ij*em ; phi_x (2 stages + dot) -> s_ij ; atomic scatter.
// ---------------------------------------------------------------------------
__global__ void message_kernel(const float* __restrict__ x,
                               const _Float16* __restrict__ h16,
                               const int* __restrict__ eidx, const uint8_t* __restrict__ emask,
                               const _Float16* Wm0, const float* bm0,
                               const _Float16* Wm1, const float* bm1,
                               const _Float16* Wm2, const float* bm2,
                               const _Float16* Wx0, const float* bx0,
                               const _Float16* Wx1, const float* bx1,
                               const float* Wx2, const float* bx2,
                               float* __restrict__ m_i, float* __restrict__ dxg,
                               float* __restrict__ degg) {
    __shared__ _Float16 sA[16 * 304];
    __shared__ _Float16 sB[16 * 144];
    __shared__ _Float16 sC[16 * 144];
    __shared__ float mF[16 * 128];
    __shared__ int sSrc[16], sDst[16];
    __shared__ float sEm[16], sXi[16][3], sXj[16][3], sD[16], sS[16];

    const int tiles = Mc / 16;
    int b  = blockIdx.x / tiles;
    int e0 = (blockIdx.x % tiles) * 16;
    int tid = threadIdx.x;

    if (tid < 16) {
        int e = e0 + tid;
        int s = eidx[(size_t)b * 2 * Mc + e];
        int d = eidx[(size_t)b * 2 * Mc + Mc + e];
        sSrc[tid] = s; sDst[tid] = d;
        sEm[tid] = emask[(size_t)b * Mc + e] ? 1.0f : 0.0f;
        float dd = 0.0f;
#pragma unroll
        for (int a = 0; a < 3; ++a) {
            float xi = x[((size_t)b * Nc + s) * 3 + a];
            float xj = x[((size_t)b * Nc + d) * 3 + a];
            sXi[tid][a] = xi; sXj[tid][a] = xj;
            float t = xi - xj; dd += t * t;
        }
        sD[tid] = dd;
    }
    __syncthreads();
    // Async gather: A tile cols [0..255] = h16[src] | h16[dst] (16B chunks).
#pragma unroll
    for (int k = 0; k < 4; ++k) {
        int idx = tid + k * 128;       // 0..511
        int region = idx >> 8;         // 0 = h_i, 1 = h_j
        int rem = idx & 255;
        int r = rem >> 4, c = rem & 15;
        int node = region ? sDst[r] : sSrc[r];
        async_copy_b128(h16 + ((size_t)b * Nc + node) * 128 + c * 8,
                        &sA[r * 304 + region * 128 + c * 8]);
    }
    // Scalar tail: col 256 = d_ij, cols 257..287 = 0.
    for (int idx = tid; idx < 16 * 32; idx += 128) {
        int r = idx >> 5, c = (idx & 31) + 256;
        sA[r * 304 + c] = (c == 256) ? (_Float16)sD[r] : (_Float16)0.0f;
    }
    wait_async();
    __syncthreads();
    gemm_f16(sA, 304, 9, Wm0, 288, bm0, true, sB, 144);  __syncthreads();
    gemm_f16(sB, 144, 4, Wm1, 128, bm1, true, sC, 144);  __syncthreads();
    gemm_f32(sC, 144, 4, Wm2, 128, bm2, false, sEm, mF, 128, sB, 144); __syncthreads();
    // phi_x on masked m_ij
    gemm_f16(sB, 144, 4, Wx0, 128, bx0, true, sC, 144);  __syncthreads();
    gemm_f16(sC, 144, 4, Wx1, 128, bx1, true, sB, 144);  __syncthreads();
    if (tid < 16) {
        float acc = bx2[0];
        for (int c = 0; c < 128; ++c) acc += (float)sB[tid * 144 + c] * Wx2[c];
        sS[tid] = acc;
    }
    __syncthreads();
    // scatter m_ij -> m_i[dst]
    for (int idx = tid; idx < 2048; idx += 128) {
        int r = idx >> 7, c = idx & 127;
        atomicAdd(&m_i[((size_t)b * Nc + sDst[r]) * 128 + c], mF[r * 128 + c]);
    }
    if (tid < 48) {
        int r = tid / 3, a = tid % 3;
        float delta = (sXi[r][a] - sXj[r][a]) * sS[r] * sEm[r];
        atomicAdd(&dxg[((size_t)b * Nc + sDst[r]) * 3 + a], delta);
    }
    if (tid < 16) atomicAdd(&degg[(size_t)b * Nc + sDst[tid]], sEm[tid]);
}

// ---------------------------------------------------------------------------
// Node update: h = LN(h + phi_h([h, m_i])) * nm ; x = (x + dx/max(deg,1)) * nm
// Writes both f32 h and the f16 mirror.
// ---------------------------------------------------------------------------
__global__ void update_kernel(float* __restrict__ h, _Float16* __restrict__ h16,
                              const float* __restrict__ m_i,
                              float* __restrict__ x, const float* __restrict__ dxg,
                              const float* __restrict__ degg,
                              const uint8_t* __restrict__ enm, const uint8_t* __restrict__ fnm,
                              const _Float16* Wh0, const float* bh0,
                              const _Float16* Wh1, const float* bh1,
                              const _Float16* Wh2, const float* bh2,
                              const float* lng, const float* lnb) {
    const int tiles = (Nc + 15) / 16;
    __shared__ _Float16 sA[16 * 272];
    __shared__ _Float16 sB[16 * 144];
    __shared__ _Float16 sC[16 * 144];
    __shared__ float hn[16 * 128];
    __shared__ float hold[16 * 128];
    int b  = blockIdx.x / tiles;
    int n0 = (blockIdx.x % tiles) * 16;
    int valid = Nc - n0; if (valid > 16) valid = 16;
    int tid = threadIdx.x;

    for (int idx = tid; idx < 16 * 256; idx += 128) {
        int r = idx / 256, c = idx % 256;
        float v = 0.0f;
        if (r < valid) {
            if (c < 128) { v = h[((size_t)b * Nc + n0 + r) * 128 + c]; hold[r * 128 + c] = v; }
            else v = m_i[((size_t)b * Nc + n0 + r) * 128 + (c - 128)];
        } else if (c < 128) hold[r * 128 + c] = 0.0f;
        sA[r * 272 + c] = (_Float16)v;
    }
    __syncthreads();
    gemm_f16(sA, 272, 8, Wh0, 256, bh0, true, sB, 144); __syncthreads();
    gemm_f16(sB, 144, 4, Wh1, 128, bh1, true, sC, 144); __syncthreads();
    gemm_f32(sC, 144, 4, Wh2, 128, bh2, false, nullptr, hn, 128, nullptr, 0);
    __syncthreads();
    if (tid < 16 && tid < valid) {
        int n = n0 + tid;
        float nm = (n < Ec) ? (enm[(size_t)b * Ec + n] ? 1.0f : 0.0f)
                            : (fnm[(size_t)b * Fc + (n - Ec)] ? 1.0f : 0.0f);
        float mu = 0.0f, var = 0.0f;
        for (int c = 0; c < 128; ++c) mu += hold[tid * 128 + c] + hn[tid * 128 + c];
        mu *= (1.0f / 128.0f);
        for (int c = 0; c < 128; ++c) {
            float v = hold[tid * 128 + c] + hn[tid * 128 + c] - mu;
            var += v * v;
        }
        var *= (1.0f / 128.0f);
        float inv = rsqrtf(var + 1e-5f);
        for (int c = 0; c < 128; ++c) {
            float v = hold[tid * 128 + c] + hn[tid * 128 + c];
            float o = ((v - mu) * inv * lng[c] + lnb[c]) * nm;
            h[((size_t)b * Nc + n) * 128 + c]   = o;
            h16[((size_t)b * Nc + n) * 128 + c] = (_Float16)o;
        }
        float dg = fmaxf(degg[(size_t)b * Nc + n], 1.0f);
        float rdg = __builtin_amdgcn_rcpf(dg);
#pragma unroll
        for (int a = 0; a < 3; ++a) {
            size_t xi = ((size_t)b * Nc + n) * 3 + a;
            x[xi] = (x[xi] + dxg[xi] * rdg) * nm;
        }
    }
}

// ---------------------------------------------------------------------------
// Heads + pools (edges or faces): logits + masked pooled-feature accumulation.
// h tile streamed from the f16 mirror via async-to-LDS.
// ---------------------------------------------------------------------------
__global__ void head_pool_kernel(const _Float16* __restrict__ h16, int rows, int tiles,
                                 int nodeOffset,
                                 const uint8_t* __restrict__ cand,
                                 const uint8_t* __restrict__ nmask,
                                 const _Float16* Wh0, const float* bh0,
                                 const _Float16* Wh1, const float* bh1,
                                 const float* Wh2v, const float* bh2,
                                 const _Float16* Wp0, const float* bp0,
                                 const _Float16* Wp1, const float* bp1,
                                 float* __restrict__ d_out, size_t logits2_off,
                                 float* __restrict__ pool_acc, float* __restrict__ den_acc) {
    __shared__ _Float16 hs[16 * 144];
    __shared__ _Float16 sB[16 * 144];
    __shared__ _Float16 sC[16 * 144];
    __shared__ float pF[16 * 128];
    int b  = blockIdx.x / tiles;
    int r0 = (blockIdx.x % tiles) * 16;
    int valid = rows - r0; if (valid > 16) valid = 16;
    int tid = threadIdx.x;

#pragma unroll
    for (int k = 0; k < 2; ++k) {
        int idx = tid + k * 128;       // 0..255 (row, 16B chunk)
        int r = idx >> 4, c = idx & 15;
        if (r < valid) {
            async_copy_b128(h16 + ((size_t)b * Nc + nodeOffset + r0 + r) * 128 + c * 8,
                            &hs[r * 144 + c * 8]);
        } else {
#pragma unroll
            for (int t2 = 0; t2 < 8; ++t2) hs[r * 144 + c * 8 + t2] = (_Float16)0.0f;
        }
    }
    wait_async();
    __syncthreads();
    // head: 128->128 silu -> 128->128 silu -> dot
    gemm_f16(hs, 144, 4, Wh0, 128, bh0, true, sB, 144); __syncthreads();
    gemm_f16(sB, 144, 4, Wh1, 128, bh1, true, sC, 144); __syncthreads();
    if (tid < valid) {
        float acc = bh2[0];
        for (int c = 0; c < 128; ++c) acc += (float)sC[tid * 144 + c] * Wh2v[c];
        int n = r0 + tid;
        float logit = (cand[(size_t)b * rows + n] != 0) ? acc : NEG_INF;
        d_out[(size_t)b * Nc + nodeOffset + n] = logit;
        d_out[logits2_off + (size_t)b * rows + n] = logit;
    }
    __syncthreads();
    // pool: 128->128 silu -> 128->128
    gemm_f16(hs, 144, 4, Wp0, 128, bp0, true, sB, 144); __syncthreads();
    gemm_f32(sB, 144, 4, Wp1, 128, bp1, false, nullptr, pF, 128, nullptr, 0);
    __syncthreads();
    for (int idx = tid; idx < 2048; idx += 128) {
        int r = idx >> 7, c = idx & 127;
        if (r < valid && nmask[(size_t)b * rows + r0 + r])
            atomicAdd(&pool_acc[b * 128 + c], pF[r * 128 + c]);
    }
    if (tid < valid && nmask[(size_t)b * rows + r0 + tid])
        atomicAdd(&den_acc[b], 1.0f);
}

// ---------------------------------------------------------------------------
// Value head (tiny: B=8 rows) — plain VALU, f32 weights straight from d_in.
// ---------------------------------------------------------------------------
__global__ void value_kernel(const float* pe_acc, const float* pf_acc,
                             const float* den_e, const float* den_f,
                             const float* Wv0, const float* bv0,
                             const float* Wv1, const float* bv1,
                             const float* Wv2, const float* bv2,
                             float* __restrict__ out) {
    __shared__ float in[256], t1[128], t2[128], red[128];
    int b = blockIdx.x, t = threadIdx.x;
    float rde = __builtin_amdgcn_rcpf(fmaxf(den_e[b], 1.0f));
    float rdf = __builtin_amdgcn_rcpf(fmaxf(den_f[b], 1.0f));
    in[t]       = pe_acc[b * 128 + t] * rde;
    in[128 + t] = pf_acc[b * 128 + t] * rdf;
    __syncthreads();
    float acc = bv0[t];
    for (int k = 0; k < 256; ++k) acc += in[k] * Wv0[(size_t)k * 128 + t];
    t1[t] = siluf(acc);
    __syncthreads();
    acc = bv1[t];
    for (int k = 0; k < 128; ++k) acc += t1[k] * Wv1[(size_t)k * 128 + t];
    t2[t] = siluf(acc);
    __syncthreads();
    red[t] = t2[t] * Wv2[t];
    __syncthreads();
    for (int s = 64; s > 0; s >>= 1) { if (t < s) red[t] += red[t + s]; __syncthreads(); }
    if (t == 0) out[b] = red[0] + bv2[0];
}

// ---------------------------------------------------------------------------
// Host launcher. Param flattening assumed jax-tree order (dict keys sorted):
// top: x, ef, ff, eidx, enm, fnm, em, ecm, fcm, then params:
//   edge_embed, edge_head, edge_pool, face_embed, face_head, face_pool,
//   layers[4]{ln(g,b), phi_h(3), phi_m(3), phi_x(3)}, value(3)
// ---------------------------------------------------------------------------
extern "C" void kernel_launch(void* const* d_in, const int* in_sizes, int n_in,
                              void* d_out, int out_size, void* d_ws, size_t ws_size,
                              hipStream_t stream) {
    (void)in_sizes; (void)n_in; (void)out_size; (void)ws_size;
    const float*   x_in = (const float*)d_in[0];
    const float*   ef   = (const float*)d_in[1];
    const float*   ff   = (const float*)d_in[2];
    const int*     eidx = (const int*)d_in[3];
    const uint8_t* enm  = (const uint8_t*)d_in[4];
    const uint8_t* fnm  = (const uint8_t*)d_in[5];
    const uint8_t* em   = (const uint8_t*)d_in[6];
    const uint8_t* ecm  = (const uint8_t*)d_in[7];
    const uint8_t* fcm  = (const uint8_t*)d_in[8];
    float* out = (float*)d_out;

    // bump allocator over d_ws
    char* ws = (char*)d_ws;
    size_t off = 0;
    auto alloc = [&](size_t bytes) -> void* {
        off = (off + 255) & ~(size_t)255;
        void* p = ws + off;
        off += bytes;
        return p;
    };

    float*    h     = (float*)alloc((size_t)Bc * Nc * DH * 4);
    _Float16* h16   = (_Float16*)alloc((size_t)Bc * Nc * DH * 2);
    float*    m_i   = (float*)alloc((size_t)Bc * Nc * DH * 4);
    float*    x_cur = (float*)alloc((size_t)Bc * Nc * 3 * 4);
    float*    dxg   = (float*)alloc((size_t)Bc * Nc * 3 * 4);
    float*    degg  = (float*)alloc((size_t)Bc * Nc * 4);
    float*    pe    = (float*)alloc((size_t)Bc * 128 * 4);
    float*    pf    = (float*)alloc((size_t)Bc * 128 * 4);
    float*    dene  = (float*)alloc((size_t)Bc * 4);
    float*    denf  = (float*)alloc((size_t)Bc * 4);

    auto conv = [&](int idx, int K, int Nout) -> _Float16* {
        int Kpad = (K + 31) & ~31;
        _Float16* dst = (_Float16*)alloc((size_t)Kpad * Nout * 2);
        int total = Kpad * Nout;
        convert_w<<<(total + 255) / 256, 256, 0, stream>>>(
            (const float*)d_in[idx], dst, K, Kpad, Nout);
        return dst;
    };

    // embeds / heads / pools
    _Float16* we0 = conv(9, 16, 128);   _Float16* we1 = conv(11, 128, 128);
    _Float16* eh0 = conv(13, 128, 128); _Float16* eh1 = conv(15, 128, 128);
    _Float16* ep0 = conv(19, 128, 128); _Float16* ep1 = conv(21, 128, 128);
    _Float16* wf0 = conv(23, 16, 128);  _Float16* wf1 = conv(25, 128, 128);
    _Float16* fh0 = conv(27, 128, 128); _Float16* fh1 = conv(29, 128, 128);
    _Float16* fp0 = conv(33, 128, 128); _Float16* fp1 = conv(35, 128, 128);

    // per-layer weights
    _Float16 *Lh0[Lc], *Lh1[Lc], *Lh2[Lc], *Lm0[Lc], *Lm1[Lc], *Lm2[Lc], *Lx0[Lc], *Lx1[Lc];
    int base[Lc];
    for (int l = 0; l < Lc; ++l) {
        base[l] = 37 + 20 * l;
        Lh0[l] = conv(base[l] + 2, 256, 128);
        Lh1[l] = conv(base[l] + 4, 128, 128);
        Lh2[l] = conv(base[l] + 6, 128, 128);
        Lm0[l] = conv(base[l] + 8, 257, 128);
        Lm1[l] = conv(base[l] + 10, 128, 128);
        Lm2[l] = conv(base[l] + 12, 128, 128);
        Lx0[l] = conv(base[l] + 14, 128, 128);
        Lx1[l] = conv(base[l] + 16, 128, 128);
    }
    const int vb = 37 + 20 * Lc;  // value MLP base = 117

    // working copy of x
    hipMemcpyAsync(x_cur, x_in, (size_t)Bc * Nc * 3 * 4, hipMemcpyDeviceToDevice, stream);

    // embeddings -> h (+ f16 mirror)
    const int tilesE = (Ec + 15) / 16, tilesF = (Fc + 15) / 16, tilesN = (Nc + 15) / 16;
    embed_kernel<<<Bc * tilesE, 128, 0, stream>>>(ef, Ec, tilesE, 0,
        we0, (const float*)d_in[10], we1, (const float*)d_in[12], h, h16);
    embed_kernel<<<Bc * tilesF, 128, 0, stream>>>(ff, Fc, tilesF, Ec,
        wf0, (const float*)d_in[24], wf1, (const float*)d_in[26], h, h16);

    // EGNN layers
    for (int l = 0; l < Lc; ++l) {
        hipMemsetAsync(m_i, 0, (size_t)Bc * Nc * DH * 4, stream);
        hipMemsetAsync(dxg, 0, (size_t)Bc * Nc * 3 * 4, stream);
        hipMemsetAsync(degg, 0, (size_t)Bc * Nc * 4, stream);
        message_kernel<<<Bc * (Mc / 16), 128, 0, stream>>>(
            x_cur, h16, eidx, em,
            Lm0[l], (const float*)d_in[base[l] + 9],
            Lm1[l], (const float*)d_in[base[l] + 11],
            Lm2[l], (const float*)d_in[base[l] + 13],
            Lx0[l], (const float*)d_in[base[l] + 15],
            Lx1[l], (const float*)d_in[base[l] + 17],
            (const float*)d_in[base[l] + 18], (const float*)d_in[base[l] + 19],
            m_i, dxg, degg);
        update_kernel<<<Bc * tilesN, 128, 0, stream>>>(
            h, h16, m_i, x_cur, dxg, degg, enm, fnm,
            Lh0[l], (const float*)d_in[base[l] + 3],
            Lh1[l], (const float*)d_in[base[l] + 5],
            Lh2[l], (const float*)d_in[base[l] + 7],
            (const float*)d_in[base[l] + 0], (const float*)d_in[base[l] + 1]);
    }

    // heads + pools
    hipMemsetAsync(pe, 0, (size_t)Bc * 128 * 4, stream);
    hipMemsetAsync(pf, 0, (size_t)Bc * 128 * 4, stream);
    hipMemsetAsync(dene, 0, (size_t)Bc * 4, stream);
    hipMemsetAsync(denf, 0, (size_t)Bc * 4, stream);
    const size_t edge2Off = (size_t)Bc * Nc;
    const size_t face2Off = edge2Off + (size_t)Bc * Ec;
    const size_t valueOff = face2Off + (size_t)Bc * Fc;
    head_pool_kernel<<<Bc * tilesE, 128, 0, stream>>>(
        h16, Ec, tilesE, 0, ecm, enm,
        eh0, (const float*)d_in[14], eh1, (const float*)d_in[16],
        (const float*)d_in[17], (const float*)d_in[18],
        ep0, (const float*)d_in[20], ep1, (const float*)d_in[22],
        out, edge2Off, pe, dene);
    head_pool_kernel<<<Bc * tilesF, 128, 0, stream>>>(
        h16, Fc, tilesF, Ec, fcm, fnm,
        fh0, (const float*)d_in[28], fh1, (const float*)d_in[30],
        (const float*)d_in[31], (const float*)d_in[32],
        fp0, (const float*)d_in[34], fp1, (const float*)d_in[36],
        out, face2Off, pf, denf);

    value_kernel<<<Bc, 128, 0, stream>>>(
        pe, pf, dene, denf,
        (const float*)d_in[vb + 0], (const float*)d_in[vb + 1],
        (const float*)d_in[vb + 2], (const float*)d_in[vb + 3],
        (const float*)d_in[vb + 4], (const float*)d_in[vb + 5],
        out + valueOff);
}
